// Head_90632399880778
// MI455X (gfx1250) — matrix operand order repaired and checked by reading, not compiled, not run on hardware
//
#include <hip/hip_runtime.h>
#include <hip/hip_bf16.h>

typedef _Float16 f16;
typedef __attribute__((ext_vector_type(16))) _Float16 v16h;
typedef __attribute__((ext_vector_type(8)))  _Float16 v8h;
typedef __attribute__((ext_vector_type(4)))  _Float16 v4h;
typedef __attribute__((ext_vector_type(4)))  float    v4f;
typedef __attribute__((ext_vector_type(8)))  float    v8f;

#define B_BATCH 512
#define T_SEQ   142
#define TP      144      // T padded to 9 tiles of 16 (Q, K, S rows)
#define TPV     160      // T padded to 5 tiles of 32 (V rows / P cols)
#define C_DIM   512
#define H_DIM   64
#define S_STRIDE 148     // f32 row stride for S (>=144, non-multiple of 64)
#define P_STRIDE 160     // f16 row stride for P (exactly 5*32)
#define NWAVES  9
#define NTHREADS (NWAVES * 32)
#define KB_STEPS (C_DIM / 32)   // 16 K-chunks in projection
#define XTASKS  (TP * 8)        // 1152 = 4 * NTHREADS exactly
#define WTASKS  (3 * 64 * 8)    // 1536 (< 6 * NTHREADS)

// ---- LDS byte offsets (one arena; double-buffered staging aliases S) ----
#define OFF_Q  0                                   // Qs  [144][64] f16 = 18432 (k-contiguous rows)
#define OFF_K  (OFF_Q + TP * H_DIM * 2)            // Ks  [144][64] f16 = 18432 (k-contiguous rows)
#define OFF_V  (OFF_K + TP * H_DIM * 2)            // Vt  [64][160] f16 = 20480 (TRANSPOSED: [n][k])
#define OFF_S  (OFF_V + H_DIM * TPV * 2)           // Ss  [144][148] f32 = 85248
#define OFF_P  (OFF_S + TP * S_STRIDE * 4)         // Ps  [144][160] f16 = 46080
#define LDS_BYTES (OFF_P + TP * P_STRIDE * 2)      // 188672 B total (< 320KB WGP LDS)
#define OFF_XST0 (OFF_S)                           // Xst [144][32] f16 = 9216  (phase A)
#define OFF_XST1 (OFF_XST0 + TP * 32 * 2)
#define OFF_WST0 (OFF_XST1 + TP * 32 * 2)          // Wt  [3][64][32] f16 = 12288 (phase A)
#define OFF_WST1 (OFF_WST0 + 3 * 64 * 32 * 2)     // end: OFF_S + 43008 <= OFF_S + 85248

// concat two contiguous 16B LDS chunks into one 16-half fragment
__device__ __forceinline__ v16h frag2(const f16* p0, const f16* p1) {
    v8h lo = *(const v8h*)p0;
    v8h hi = *(const v8h*)p1;
    return __builtin_shufflevector(lo, hi, 0,1,2,3,4,5,6,7,8,9,10,11,12,13,14,15);
}
// A-fragment per ISA 16-bit A layout: chunks [kg*8, +8) and [16+kg*8, +8)
__device__ __forceinline__ v16h fragA(const f16* row, int kg, int ks32) {
    return frag2(row + ks32 + kg * 8, row + ks32 + 16 + kg * 8);
}
// B-fragment matching A layout: chunks [kg*8, +8) and [16+kg*8, +8)
__device__ __forceinline__ v16h fragB(const f16* col, int kg, int ks32) {
    return frag2(col + ks32 + kg * 8, col + ks32 + 16 + kg * 8);
}
__device__ __forceinline__ v4h cvt4(v4f f) {
    return v4h{(f16)f.x, (f16)f.y, (f16)f.z, (f16)f.w};
}

__global__ __launch_bounds__(NTHREADS)
void fused_head_attn(const float* __restrict__ x,
                     const float* __restrict__ Wq,
                     const float* __restrict__ Wk,
                     const float* __restrict__ Wv,
                     float* __restrict__ out)
{
    __shared__ __attribute__((aligned(128))) char lds[LDS_BYTES];
    f16*   Qs   = (f16*)(lds + OFF_Q);
    f16*   Ks   = (f16*)(lds + OFF_K);
    f16*   Vt   = (f16*)(lds + OFF_V);   // Vt[n][k] : k = key index, 160 per row
    float* Ss   = (float*)(lds + OFF_S);
    f16*   Ps   = (f16*)(lds + OFF_P);
    f16*   Xst0 = (f16*)(lds + OFF_XST0);
    f16*   Xst1 = (f16*)(lds + OFF_XST1);
    f16*   Wst0 = (f16*)(lds + OFF_WST0);
    f16*   Wst1 = (f16*)(lds + OFF_WST1);

    const int b    = blockIdx.x;
    const int tid  = threadIdx.x;
    const int wave = tid >> 5;
    const int lane = tid & 31;
    const int ln   = lane & 15;   // M (A-frag) or N (B/C-frag) index within 16-tile
    const int kg   = lane >> 4;   // lane group 0/1

    const float* xb = x + (size_t)b * T_SEQ * C_DIM;
    const int mrow  = wave * 16;  // this wave's M-tile base row

    // zero V padding keys [144,160) in transposed layout
    for (int i = tid; i < H_DIM * (TPV - TP); i += NTHREADS) {
        int n = i >> 4, k = TP + (i & 15);
        Vt[n * TPV + k] = (f16)0.f;
    }

    // ========== Phase A: Q/K/V = x @ W, double-buffered software pipeline ===
    v8f accQ[4] = {}, accK[4] = {}, accV[4] = {};

    // prologue: stage kb=0 into buffer 0
    {
        #pragma unroll
        for (int u = 0; u < 4; ++u) {
            int j = tid + u * NTHREADS;               // covers XTASKS exactly
            int t = j >> 3, cq = (j & 7) * 4;
            v4f f4 = (t < T_SEQ) ? *(const v4f*)(xb + t * C_DIM + cq)
                                 : v4f{0.f, 0.f, 0.f, 0.f};
            *(v4h*)(Xst0 + t * 32 + cq) = cvt4(f4);
        }
        #pragma unroll
        for (int u = 0; u < 6; ++u) {
            int j = tid + u * NTHREADS;
            if (j < WTASKS) {                          // cuts at a wave boundary
                int m = j >> 9, n = (j >> 3) & 63, k4 = (j & 7) * 4;
                const float* Wm = (m == 0) ? Wq : (m == 1) ? Wk : Wv;
                const float* src = Wm + k4 * H_DIM + n;
                *(v4h*)(Wst0 + m * 2048 + n * 32 + k4) =
                    cvt4(v4f{src[0], src[H_DIM], src[2 * H_DIM], src[3 * H_DIM]});
            }
        }
    }
    __syncthreads();

    for (int kb = 0; kb < KB_STEPS; ++kb) {
        f16* Xc = (kb & 1) ? Xst1 : Xst0;
        f16* Wc = (kb & 1) ? Wst1 : Wst0;
        f16* Xn = (kb & 1) ? Xst0 : Xst1;
        f16* Wn = (kb & 1) ? Wst0 : Wst1;
        const bool pf = (kb + 1 < KB_STEPS);
        const int  kn = (kb + 1) * 32;

        // (1) issue global loads for step kb+1 (LOADcnt in flight during WMMAs)
        v4f xr[4], wr[6];
        if (pf) {
            #pragma unroll
            for (int u = 0; u < 4; ++u) {
                int j = tid + u * NTHREADS;
                int t = j >> 3, cq = (j & 7) * 4;
                xr[u] = (t < T_SEQ) ? *(const v4f*)(xb + t * C_DIM + kn + cq)
                                    : v4f{0.f, 0.f, 0.f, 0.f};
            }
            #pragma unroll
            for (int u = 0; u < 6; ++u) {
                int j = tid + u * NTHREADS;
                if (j < WTASKS) {
                    int m = j >> 9, n = (j >> 3) & 63, k4 = (j & 7) * 4;
                    const float* Wm = (m == 0) ? Wq : (m == 1) ? Wk : Wv;
                    const float* src = Wm + (kn + k4) * H_DIM + n;
                    wr[u] = v4f{src[0], src[H_DIM], src[2 * H_DIM], src[3 * H_DIM]};
                }
            }
        }

        // (2) 12 WMMAs for step kb (waits touch DScnt only; global loads keep flying)
        v16h a = fragA(Xc + (mrow + ln) * 32, kg, 0);
        #pragma unroll
        for (int nt = 0; nt < 4; ++nt) {
            int col = nt * 16 + ln;
            v16h bq = fragB(Wc + 0 * 2048 + col * 32, kg, 0);
            v16h bk = fragB(Wc + 1 * 2048 + col * 32, kg, 0);
            v16h bv = fragB(Wc + 2 * 2048 + col * 32, kg, 0);
            accQ[nt] = __builtin_amdgcn_wmma_f32_16x16x32_f16(false, a, false, bq, (short)0, accQ[nt], false, false);
            accK[nt] = __builtin_amdgcn_wmma_f32_16x16x32_f16(false, a, false, bk, (short)0, accK[nt], false, false);
            accV[nt] = __builtin_amdgcn_wmma_f32_16x16x32_f16(false, a, false, bv, (short)0, accV[nt], false, false);
        }

        // (3) convert + store prefetched data into the alternate buffer
        if (pf) {
            #pragma unroll
            for (int u = 0; u < 4; ++u) {
                int j = tid + u * NTHREADS;
                int t = j >> 3, cq = (j & 7) * 4;
                *(v4h*)(Xn + t * 32 + cq) = cvt4(xr[u]);
            }
            #pragma unroll
            for (int u = 0; u < 6; ++u) {
                int j = tid + u * NTHREADS;
                if (j < WTASKS) {
                    int m = j >> 9, n = (j >> 3) & 63, k4 = (j & 7) * 4;
                    *(v4h*)(Wn + m * 2048 + n * 32 + k4) = cvt4(wr[u]);
                }
            }
        }
        __syncthreads();   // single barrier per pipeline stage
    }

    // C/D layout: VGPR r, lane -> row (r + 8*kg), col ln.  V stored transposed.
    #pragma unroll
    for (int nt = 0; nt < 4; ++nt) {
        #pragma unroll
        for (int r = 0; r < 8; ++r) {
            int row = mrow + r + kg * 8;
            int col = nt * 16 + ln;
            Qs[row * H_DIM + col] = (f16)accQ[nt][r];
            Ks[row * H_DIM + col] = (f16)accK[nt][r];
            Vt[col * TPV + row]   = (f16)accV[nt][r];
        }
    }
    __syncthreads();

    // ================= Phase B1: S = scale * Q @ K^T (causal tiles only) ====
    const float scale = 0.044194173824159216f;   // 512^-0.5 (embed dim, as in ref)
    for (int nt = 0; nt <= wave; ++nt) {         // uniform per wave: EXEC stays full
        v8f acc = {};
        #pragma unroll
        for (int ks = 0; ks < 2; ++ks) {         // H=64 -> two K=32 steps
            v16h a  = fragA(Qs + (mrow + ln) * H_DIM, kg, ks * 32);
            v16h bb = fragB(Ks + (nt * 16 + ln) * H_DIM, kg, ks * 32); // B=K^T: rows k-contig
            acc = __builtin_amdgcn_wmma_f32_16x16x32_f16(false, a, false, bb, (short)0, acc, false, false);
        }
        #pragma unroll
        for (int r = 0; r < 8; ++r) {
            int row = mrow + r + kg * 8;
            Ss[row * S_STRIDE + nt * 16 + ln] = acc[r] * scale;
        }
    }
    __syncthreads();

    // ================= Phase B2: causal softmax rows -> P (f16) =============
    if (tid < TP) {
        const int row = tid;
        float m = -3.0e38f;
        for (int c = 0; c <= row && c < TP; ++c)
            m = fmaxf(m, Ss[row * S_STRIDE + c]);
        float sum = 0.f;
        for (int c = 0; c <= row && c < TP; ++c)
            sum += __expf(Ss[row * S_STRIDE + c] - m);
        const float inv = 1.0f / sum;
        int c = 0;
        for (; c <= row && c < TP; ++c)
            Ps[row * P_STRIDE + c] = (f16)(__expf(Ss[row * S_STRIDE + c] - m) * inv);
        for (; c < P_STRIDE; ++c)
            Ps[row * P_STRIDE + c] = (f16)0.f;    // masked + padded cols
    }
    __syncthreads();

    // ================= Phase B3: O = P @ V, stream f32 to HBM ===============
    float* outb = out + (size_t)b * T_SEQ * H_DIM;
    #pragma unroll
    for (int nt = 0; nt < 4; ++nt) {
        v8f acc = {};
        #pragma unroll
        for (int ks = 0; ks < TPV / 32; ++ks) {   // 5 K=32 steps over keys
            v16h a  = fragA(Ps + (mrow + ln) * P_STRIDE, kg, ks * 32);
            v16h bb = fragB(Vt + (nt * 16 + ln) * TPV, kg, ks * 32);   // Vt[n][k] k-contig
            acc = __builtin_amdgcn_wmma_f32_16x16x32_f16(false, a, false, bb, (short)0, acc, false, false);
        }
        #pragma unroll
        for (int r = 0; r < 8; ++r) {
            int row = mrow + r + kg * 8;
            if (row < T_SEQ)
                outb[row * H_DIM + nt * 16 + ln] = acc[r];
        }
    }
}

extern "C" void kernel_launch(void* const* d_in, const int* in_sizes, int n_in,
                              void* d_out, int out_size, void* d_ws, size_t ws_size,
                              hipStream_t stream) {
    const float* x  = (const float*)d_in[0];
    const float* Wq = (const float*)d_in[1];
    const float* Wk = (const float*)d_in[2];
    const float* Wv = (const float*)d_in[3];
    float* out = (float*)d_out;
    fused_head_attn<<<dim3(B_BATCH), dim3(NTHREADS), 0, stream>>>(x, Wq, Wk, Wv, out);
}
